// SelectiveSSM_83932250898896
// MI455X (gfx1250) — matrix operand order repaired and checked by reading, not compiled, not run on hardware
//
#include <hip/hip_runtime.h>
#include <hip/hip_bf16.h>
#include <math.h>

#define D_MODEL 1024
#define D_STATE 16
#define DT_RANK 64
#define BATCH   2
#define SEQLEN  2048
#define M_ROWS  (BATCH * SEQLEN)          // 4096
#define K_DBL   (DT_RANK + 2 * D_STATE)   // 96
#define NCHUNK  16
#define CHUNK   (SEQLEN / NCHUNK)         // 128
#define NLANES  (BATCH * D_MODEL * D_STATE)          // 32768 (b,d,n)
#define NLANES1 (NLANES * NCHUNK)                     // 524288 (b,d,n,chunk)

typedef __attribute__((ext_vector_type(16))) __bf16    v16bf;
typedef __attribute__((ext_vector_type(8)))  float     v8f;
typedef __attribute__((ext_vector_type(4)))  unsigned  v4u;
typedef __attribute__((ext_vector_type(8)))  int       v8i;
typedef __attribute__((ext_vector_type(4)))  int       v4i;

__device__ __forceinline__ unsigned short f32_to_bf16(float f) {
    union { float f; unsigned u; } v; v.f = f;
    unsigned u = v.u;
    u += 0x7FFFu + ((u >> 16) & 1u);   // round-to-nearest-even
    return (unsigned short)(u >> 16);
}
__device__ __forceinline__ unsigned pack_bf16(float lo, float hi) {
    return (unsigned)f32_to_bf16(lo) | ((unsigned)f32_to_bf16(hi) << 16);
}

union FragBF { v16bf v; unsigned u[8]; };

// ---------------------------------------------------------------------------
// TDM: issue a 2-D tensor_load_to_lds. Descriptor per CDNA5 ISA §8.3/8.4.
// data_size = 4 bytes; LDS padding: +1 DWORD after every 64 DWORDs (rows are
// 64 elements -> LDS row stride 65 floats -> conflict-free (65n+k) mod 64).
// Must be executed by exactly one wave (TDM ignores EXEC; issues per wave).
// Toolchain here = clang-23 6-arg builtin: (v4u, v8i, v4i, v4i, v8i, i32).
// ---------------------------------------------------------------------------
__device__ __forceinline__ int rfl(unsigned x) {
    return __builtin_amdgcn_readfirstlane((int)x);
}

__device__ __forceinline__ void tdm_load_2d_f32(unsigned lds_off, const void* gptr,
                                                unsigned tensor_d0, unsigned tensor_d1,
                                                unsigned tile_d0, unsigned tile_d1,
                                                unsigned d0_stride) {
    unsigned long long ga = (unsigned long long)(size_t)gptr;
    v4u g0;
    g0.x = (unsigned)rfl(1u);                                        // count=1 (valid user D#)
    g0.y = (unsigned)rfl(lds_off);                                   // lds_addr (bytes)
    g0.z = (unsigned)rfl((unsigned)(ga & 0xFFFFFFFFu));              // global_addr[31:0]
    g0.w = (unsigned)rfl(((unsigned)((ga >> 32) & 0x1FFFFFFu)) | (2u << 30)); // addr[56:32] | type=2
    v8i g1;
    g1[0] = rfl((2u << 16) | (1u << 20) | (5u << 22));               // data_size=4B, pad_en, interval=64dw, amount=1dw
    g1[1] = rfl((tensor_d0 & 0xFFFFu) << 16);                        // tensor_dim0[15:0]
    g1[2] = rfl((tensor_d0 >> 16) | ((tensor_d1 & 0xFFFFu) << 16));  // tensor_dim0[31:16] | tensor_dim1[15:0]
    g1[3] = rfl((tensor_d1 >> 16) | (tile_d0 << 16));                // tensor_dim1[31:16] | tile_dim0
    g1[4] = rfl(tile_d1 & 0xFFFFu);                                  // tile_dim1 | tile_dim2=0
    g1[5] = rfl(d0_stride);                                          // tensor_dim0_stride[31:0]
    g1[6] = rfl(0u);                                                 // stride0[47:32] | stride1[15:0]
    g1[7] = rfl(0u);                                                 // stride1[47:16]
    v4i z4 = {0, 0, 0, 0};                                           // groups 2/3 unused (2-D tile)
    v8i z8 = {0, 0, 0, 0, 0, 0, 0, 0};                               // trailing group unused
    __builtin_amdgcn_tensor_load_to_lds(g0, g1, z4, z4, z8, 0);
}

__device__ __forceinline__ unsigned lds_offset_of(const void* p) {
    return (unsigned)(unsigned long long)(size_t)p;   // low 32 bits of flat addr = LDS offset
}

// ---------------------------------------------------------------------------
// Kernel 1: x_dbl[M,96] = x[M,1024] * W_xproj[96,1024]^T   (bf16 WMMA, f32 acc)
// Block: 128 threads (4 waves). Tile: 64 rows x 96 cols, K chunks of 32.
// ---------------------------------------------------------------------------
__global__ __launch_bounds__(128)
void xproj_gemm(const float* __restrict__ x,
                const float* __restrict__ Wx,
                float* __restrict__ x_dbl) {
    __shared__ unsigned short lAs[64 * 34];   // 64x32 bf16, stride 34
    __shared__ unsigned short lBs[96 * 34];   // 96x32 bf16

    const int tid  = threadIdx.x;
    const int wid  = tid >> 5;
    const int lane = tid & 31;
    const int ml   = lane & 15;
    const int hi   = lane >> 4;
    const int blockM = blockIdx.x * 64;

    const v8f vzero = {0.f, 0.f, 0.f, 0.f, 0.f, 0.f, 0.f, 0.f};
    v8f acc[6];
#pragma unroll
    for (int t = 0; t < 6; ++t) acc[t] = vzero;

    for (int k0 = 0; k0 < D_MODEL; k0 += 32) {
        __syncthreads();
        // A tile: 64 rows x 32 k, float4 loads + packed bf16 stores
        for (int flat = tid; flat < 64 * 8; flat += 128) {
            int r = flat >> 3, c4 = (flat & 7) * 4;
            float4 v = *(const float4*)&x[(size_t)(blockM + r) * D_MODEL + k0 + c4];
            *(unsigned*)&lAs[r * 34 + c4]     = pack_bf16(v.x, v.y);
            *(unsigned*)&lAs[r * 34 + c4 + 2] = pack_bf16(v.z, v.w);
        }
        // B tile: 96 output-cols x 32 k
        for (int flat = tid; flat < 96 * 8; flat += 128) {
            int n = flat >> 3, c4 = (flat & 7) * 4;
            float4 v = *(const float4*)&Wx[(size_t)n * D_MODEL + k0 + c4];
            *(unsigned*)&lBs[n * 34 + c4]     = pack_bf16(v.x, v.y);
            *(unsigned*)&lBs[n * 34 + c4 + 2] = pack_bf16(v.z, v.w);
        }
        __syncthreads();

        FragBF afr;   // ISA 16-bit A 16x32 layout
#pragma unroll
        for (int j = 0; j < 8; ++j) {
            int kk = 2 * j + ((j >= 4) ? 8 : 0) + hi * 8;
            afr.u[j] = *(const unsigned*)&lAs[(wid * 16 + ml) * 34 + kk];
        }
#pragma unroll
        for (int ct = 0; ct < 6; ++ct) {
            FragBF bfr;
            const unsigned short* src = &lBs[(ct * 16 + ml) * 34 + hi * 8];
#pragma unroll
            for (int j = 0; j < 8; ++j) bfr.u[j] = *(const unsigned*)&src[2 * j + ((j >= 4) ? 8 : 0)];
            acc[ct] = __builtin_amdgcn_wmma_f32_16x16x32_bf16(
                false, afr.v, false, bfr.v, (short)0, acc[ct], false, false);
        }
    }

#pragma unroll
    for (int ct = 0; ct < 6; ++ct)
#pragma unroll
        for (int r = 0; r < 8; ++r) {
            int row = blockM + wid * 16 + r + 8 * hi;
            int col = ct * 16 + ml;
            x_dbl[(size_t)row * K_DBL + col] = acc[ct][r];
        }
}

// ---------------------------------------------------------------------------
// Kernel 2: dt[M,1024] = softplus(dt_low[M,64] * W_dtproj[1024,64]^T + b)
// Operand staging via the Tensor Data Mover (async DMA to LDS, f32 tiles with
// row padding), fragments converted f32->bf16 at build time.
// Block: 128 threads (4 waves). Tile: 16 rows x 256 cols, K = 64 (2 chunks).
// ---------------------------------------------------------------------------
__global__ __launch_bounds__(128)
void dtproj_gemm(const float* __restrict__ x_dbl,
                 const float* __restrict__ Wdt,
                 const float* __restrict__ bdt,
                 float* __restrict__ dt_out) {
    __shared__ float lAf[16 * 65];    // 16x64 f32, padded rows (stride 65)
    __shared__ float lBf[256 * 65];   // 256x64 f32, padded rows

    const int tid  = threadIdx.x;
    const int wid  = tid >> 5;
    const int lane = tid & 31;
    const int ml   = lane & 15;
    const int hi   = lane >> 4;
    const int blockM = blockIdx.x * 16;
    const int blockN = blockIdx.y * 256;

    if (tid < 32) {   // wave 0 issues both TDM descriptors
        // A: 16 rows x 64 cols of x_dbl (row stride 96)
        tdm_load_2d_f32(lds_offset_of(lAf), &x_dbl[(size_t)blockM * K_DBL],
                        /*tensor_d0=*/K_DBL, /*tensor_d1=*/16,
                        /*tile_d0=*/DT_RANK, /*tile_d1=*/16,
                        /*d0_stride=*/K_DBL);
        // B: 256 rows x 64 cols of W_dtproj (contiguous, row stride 64)
        tdm_load_2d_f32(lds_offset_of(lBf), &Wdt[(size_t)blockN * DT_RANK],
                        /*tensor_d0=*/DT_RANK, /*tensor_d1=*/256,
                        /*tile_d0=*/DT_RANK, /*tile_d1=*/256,
                        /*d0_stride=*/DT_RANK);
        __builtin_amdgcn_s_wait_tensorcnt(0);
    }
    __syncthreads();

    FragBF afr[2];
#pragma unroll
    for (int kc = 0; kc < 2; ++kc)
#pragma unroll
        for (int j = 0; j < 8; ++j) {
            int kk = kc * 32 + 2 * j + ((j >= 4) ? 8 : 0) + hi * 8;
            afr[kc].u[j] = pack_bf16(lAf[ml * 65 + kk], lAf[ml * 65 + kk + 1]);
        }

    const v8f vzero = {0.f, 0.f, 0.f, 0.f, 0.f, 0.f, 0.f, 0.f};
#pragma unroll
    for (int ct = 0; ct < 4; ++ct) {
        const int n_loc = wid * 64 + ct * 16 + ml;
        v8f acc = vzero;
#pragma unroll
        for (int kc = 0; kc < 2; ++kc) {
            FragBF bfr;
            const float* src = &lBf[n_loc * 65 + kc * 32 + hi * 8];
#pragma unroll
            for (int j = 0; j < 8; ++j) bfr.u[j] = pack_bf16(src[2 * j + ((j >= 4) ? 8 : 0)], src[2 * j + ((j >= 4) ? 8 : 0) + 1]);
            acc = __builtin_amdgcn_wmma_f32_16x16x32_bf16(
                false, afr[kc].v, false, bfr.v, (short)0, acc, false, false);
        }
        const int col = blockN + n_loc;
        const float bias = bdt[col];
#pragma unroll
        for (int r = 0; r < 8; ++r) {
            int row = blockM + r + 8 * hi;
            float z = acc[r] + bias;
            float sp = fmaxf(z, 0.0f) + log1pf(__expf(-fabsf(z)));  // softplus
            dt_out[(size_t)row * D_MODEL + col] = sp;
        }
    }
}

// ---------------------------------------------------------------------------
// Chunked selective scan (diagonal linear recurrence h = dA*h + dBu):
//   pass1: per (b,d,n,chunk): local scan of 128 steps -> P = prod(dA), q = h_local
//   pass2: per (b,d,n): 16-step prefix combine -> chunk entry states H
//   pass3: per (b,d,n,chunk): rerun local scan from H, reduce y over n (shfl)
// Lane map: t = chunk*32768 + (b*1024+d)*16 + n  (16-lane n-groups per wave)
// ---------------------------------------------------------------------------
__global__ __launch_bounds__(256)
void scan_pass1(const float* __restrict__ x, const float* __restrict__ x_dbl,
                const float* __restrict__ dt, const float* __restrict__ A_log,
                float* __restrict__ Pbuf, float* __restrict__ Qbuf) {
    const int t     = blockIdx.x * 256 + threadIdx.x;     // 0..NLANES1-1
    const int n     = t & 15;
    const int ch    = t >> 4;
    const int d     = ch & (D_MODEL - 1);
    const int b     = (ch >> 10) & (BATCH - 1);
    const int chunk = ch >> 11;

    const float a = -__expf(A_log[d * D_STATE + n]);
    const size_t rowbase = (size_t)(b * SEQLEN + chunk * CHUNK);
    const float* dtp = dt    + rowbase * D_MODEL + d;
    const float* xp  = x     + rowbase * D_MODEL + d;
    const float* xdb = x_dbl + rowbase * K_DBL + DT_RANK + n;

    float h = 0.0f, P = 1.0f;
    for (int l = 0; l < CHUNK; ++l) {
        __builtin_prefetch(&dtp[(size_t)(l + 16) * D_MODEL], 0, 0);
        __builtin_prefetch(&xp[(size_t)(l + 16) * D_MODEL], 0, 0);
        float dtv = dtp[(size_t)l * D_MODEL];
        float xv  = xp[(size_t)l * D_MODEL];
        float Bv  = xdb[(size_t)l * K_DBL];
        float dA  = __expf(dtv * a);
        P *= dA;
        h = fmaf(dA, h, dtv * Bv * xv);
    }
    Pbuf[t] = P;
    Qbuf[t] = h;
}

__global__ __launch_bounds__(256)
void scan_pass2(const float* __restrict__ Pbuf, const float* __restrict__ Qbuf,
                float* __restrict__ Hbuf) {
    const int t = blockIdx.x * 256 + threadIdx.x;   // 0..NLANES-1 : (b,d,n)
    float H = 0.0f;
#pragma unroll
    for (int c = 0; c < NCHUNK; ++c) {
        const int idx = c * NLANES + t;
        Hbuf[idx] = H;                               // entry state of chunk c
        H = fmaf(Pbuf[idx], H, Qbuf[idx]);
    }
}

__global__ __launch_bounds__(256)
void scan_pass3(const float* __restrict__ x, const float* __restrict__ x_dbl,
                const float* __restrict__ dt, const float* __restrict__ A_log,
                const float* __restrict__ D_param, const float* __restrict__ Hbuf,
                float* __restrict__ y) {
    const int t     = blockIdx.x * 256 + threadIdx.x;
    const int n     = t & 15;
    const int ch    = t >> 4;
    const int d     = ch & (D_MODEL - 1);
    const int b     = (ch >> 10) & (BATCH - 1);
    const int chunk = ch >> 11;

    const float a  = -__expf(A_log[d * D_STATE + n]);
    const float Dp = D_param[d];
    const size_t rowbase = (size_t)(b * SEQLEN + chunk * CHUNK);
    const float* dtp = dt    + rowbase * D_MODEL + d;
    const float* xp  = x     + rowbase * D_MODEL + d;
    const float* xdb = x_dbl + rowbase * K_DBL + DT_RANK + n;   // B_n; C_n at +16
    float*       yp  = y     + rowbase * D_MODEL + d;

    float h = Hbuf[t];
    for (int l = 0; l < CHUNK; ++l) {
        __builtin_prefetch(&dtp[(size_t)(l + 16) * D_MODEL], 0, 0);
        __builtin_prefetch(&xp[(size_t)(l + 16) * D_MODEL], 0, 0);
        float dtv = dtp[(size_t)l * D_MODEL];
        float xv  = xp[(size_t)l * D_MODEL];
        float Bv  = xdb[(size_t)l * K_DBL];
        float Cv  = xdb[(size_t)l * K_DBL + D_STATE];
        float dA  = __expf(dtv * a);
        h = fmaf(dA, h, dtv * Bv * xv);
        float p = h * Cv;
        p += __shfl_xor(p, 1, 32);
        p += __shfl_xor(p, 2, 32);
        p += __shfl_xor(p, 4, 32);
        p += __shfl_xor(p, 8, 32);
        if (n == 0) yp[(size_t)l * D_MODEL] = fmaf(Dp, xv, p);
    }
}

// ---------------------------------------------------------------------------
extern "C" void kernel_launch(void* const* d_in, const int* in_sizes, int n_in,
                              void* d_out, int out_size, void* d_ws, size_t ws_size,
                              hipStream_t stream) {
    const float* x     = (const float*)d_in[0];
    const float* Wx    = (const float*)d_in[1];
    const float* Wdt   = (const float*)d_in[2];
    const float* bdt   = (const float*)d_in[3];
    const float* A_log = (const float*)d_in[4];
    const float* Dp    = (const float*)d_in[5];
    float* out = (float*)d_out;

    char* ws = (char*)d_ws;
    size_t off = 0;
    float* x_dbl = (float*)(ws + off); off += (size_t)M_ROWS * K_DBL * sizeof(float);   // 1.5 MB
    float* dtbuf = (float*)(ws + off); off += (size_t)M_ROWS * D_MODEL * sizeof(float); // 16 MB
    float* Pbuf  = (float*)(ws + off); off += (size_t)NLANES1 * sizeof(float);          // 2 MB
    float* Qbuf  = (float*)(ws + off); off += (size_t)NLANES1 * sizeof(float);          // 2 MB
    float* Hbuf  = (float*)(ws + off); off += (size_t)NLANES1 * sizeof(float);          // 2 MB

    xproj_gemm<<<M_ROWS / 64, 128, 0, stream>>>(x, Wx, x_dbl);
    dtproj_gemm<<<dim3(M_ROWS / 16, D_MODEL / 256), 128, 0, stream>>>(x_dbl, Wdt, bdt, dtbuf);
    scan_pass1<<<NLANES1 / 256, 256, 0, stream>>>(x, x_dbl, dtbuf, A_log, Pbuf, Qbuf);
    scan_pass2<<<NLANES / 256, 256, 0, stream>>>(Pbuf, Qbuf, Hbuf);
    scan_pass3<<<NLANES1 / 256, 256, 0, stream>>>(x, x_dbl, dtbuf, A_log, Dp, Hbuf, out);
}
